// HIRCLSMILHead_15281493639675
// MI455X (gfx1250) — matrix operand order repaired and checked by reading, not compiled, not run on hardware
//
#include <hip/hip_runtime.h>
#include <hip/hip_fp16.h>

// ---------------------------------------------------------------------------
// MI455X (gfx1250) implementation of the hierarchical-MIL head.
//
// GEMM1 (65536x1024 @ 1024x512) + GEMM2 (65536x512 @ 512x256) fused in one
// kernel via v_wmma_f32_16x16x32_bf16. Data movement uses the CDNA5 paths:
//   - Tensor Data Mover (tensor_load_to_lds) for the 2D strided A tile
//   - global_load_async_to_lds_b128 for contiguous weight tiles
// Embeddings (256MB) are read once; region_emb round-trips HBM once as f16.
// ---------------------------------------------------------------------------

typedef __attribute__((ext_vector_type(16))) __bf16 v16bf;
typedef __attribute__((ext_vector_type(8)))  float  v8f;
typedef __attribute__((ext_vector_type(4)))  unsigned int v4u;
typedef __attribute__((ext_vector_type(8)))  int    v8i;

#define DEV __device__ __forceinline__

DEV float gelu_exact(float x) { return 0.5f * x * (1.0f + erff(x * 0.70710678118654752f)); }

DEV unsigned int lds_addr_of(const void* p) {
  // LDS aperture: low 32 bits of the flat address are the LDS byte address.
  return (unsigned int)(unsigned long long)p;
}

// Async 16B global -> LDS copy (ASYNCcnt tracked, no VGPR round trip).
DEV void async_copy_b128(unsigned int lds_byte_addr, unsigned long long gaddr) {
  asm volatile("global_load_async_to_lds_b128 %0, %1, off"
               :: "v"(lds_byte_addr), "v"(gaddr) : "memory");
}
DEV void wait_asynccnt0() { asm volatile("s_wait_asynccnt 0x0" ::: "memory"); }

// ---------------------------------------------------------------------------
// Kernel 0: pack an f32 (K x N) row-major weight matrix into bf16 tiles laid
// out exactly as per-lane WMMA B fragments: tile(kb,nb) = 32x16, flat idx =
// (kb*(N/16)+nb)*512 + lane*16 + e; B frag: col N = lane%16, K = (e%8)+(lane/16)*8+(e/8)*16
// ---------------------------------------------------------------------------
__global__ void pack_b_bf16(const float* __restrict__ W, __bf16* __restrict__ out,
                            int K, int N) {
  int idx = blockIdx.x * 256 + threadIdx.x;
  if (idx >= K * N) return;
  int tile = idx >> 9;
  int r    = idx & 511;
  int lane = r >> 4;
  int e    = r & 15;
  int ntiles = N >> 4;
  int kb = tile / ntiles, nb = tile % ntiles;
  int k = kb * 32 + (e & 7) + ((lane >> 4) << 3) + ((e >> 3) << 4);
  int n = nb * 16 + (lane & 15);
  out[idx] = (__bf16)W[k * N + n];
}

// ---------------------------------------------------------------------------
// Kernel 1: fused  h = gelu(X @ W1 + b1); s = tanh(h @ Wa1 + ba1).Wa2 + ba2
// Per block: 64-patch M tile, 8 waves in a 2(M) x 4(N) grid.
// ---------------------------------------------------------------------------
__global__ __launch_bounds__(256, 1) void mil_stage1(
    const float*  __restrict__ emb,    // (65536, 1024) f32
    const __bf16* __restrict__ W1t,    // packed (1024x512) bf16 tiles
    const float*  __restrict__ b1,
    const __bf16* __restrict__ Wa1t,   // packed (512x256) bf16 tiles
    const float*  __restrict__ ba1,
    const float*  __restrict__ Wa2,
    const float*  __restrict__ ba2,
    _Float16*     __restrict__ hout,   // (65536, 512) f16
    float*        __restrict__ scores) // (65536)
{
  // phase1: { A 64 rows x 36 dwords f32 (TDM-padded, 9216B), B 32x512 bf16 (32KB) }
  // phase2: { H 64x528 bf16 (66KB), W2 8192 bf16 (16KB) }  -- aliased
  __shared__ __align__(16) char smem[64 * 528 * 2 + 8192 * 2]; // 83968 B
  __shared__ float s_score[64];

  float*  Asf = (float*)smem;                   // 64 x 36 f32 (stride 144B)
  __bf16* Bs  = (__bf16*)(smem + 9216);         // 32 x 512 tile-layout
  __bf16* Hs  = (__bf16*)smem;                  // 64 x 528
  __bf16* W2s = (__bf16*)(smem + 64 * 528 * 2); // 16 tiles x 512

  const int tid  = threadIdx.x;
  const int lane = tid & 31;
  const int wid  = tid >> 5;
  const int wm   = wid >> 2;   // 0..1  (M wave)
  const int wn   = wid & 3;    // 0..3  (N wave)
  const int half = lane >> 4;
  const int l16  = lane & 15;
  const int patch0 = blockIdx.x * 64;
  const int tid_u = __builtin_amdgcn_readfirstlane(threadIdx.x); // uniform branch key

  const unsigned int as_lds  = lds_addr_of(Asf);
  const unsigned int bs_lds  = lds_addr_of(Bs);
  const unsigned int w2s_lds = lds_addr_of(W2s);
  const unsigned long long w1t_ga  = (unsigned long long)W1t;
  const unsigned long long wa1t_ga = (unsigned long long)Wa1t;

  // TDM descriptor (2D, constant parts). Fields per CDNA5 ISA 8.3/8.4.
  // group0: w0 = count=1 ; w1 = lds_addr ; w2/w3 = global_addr ; type=2
  // group1: data_size=4B(2), pad_enable, pad_interval=32dw(code4),
  //         pad_amount=4dw(code3); tile = 32 x 64; dim0_stride = 1024 elems.
  const unsigned int TD0 = 1u << 30, TD1 = 1u << 30; // huge dims: tiles always in-bounds
  v8i g1;
  g1[0] = (int)((2u << 16) | (1u << 20) | (4u << 22) | (3u << 25));
  g1[1] = (int)((TD0 & 0xFFFFu) << 16);
  g1[2] = (int)((TD0 >> 16) | ((TD1 & 0xFFFFu) << 16));
  g1[3] = (int)((TD1 >> 16) | (32u << 16));          // tile_dim0 = 32
  g1[4] = 64;                                        // tile_dim1 = 64, tile_dim2 = 0
  g1[5] = 1024;                                      // tensor_dim0_stride lo
  g1[6] = 0;                                         // stride hi / dim1_stride (unused 2D)
  g1[7] = 0;

  const v8f vzero = {0.f, 0.f, 0.f, 0.f, 0.f, 0.f, 0.f, 0.f};

  // ----------------- phase 1: X @ W1 -----------------
  v8f acc[2][8];
  #pragma unroll
  for (int m = 0; m < 2; ++m)
    #pragma unroll
    for (int n = 0; n < 8; ++n) acc[m][n] = vzero;

  for (int kb = 0; kb < 32; ++kb) {
    __syncthreads();
    if (tid_u < 32) { // wave 0 only -- scalar branch (TDM ignores EXEC)
      unsigned long long ga = (unsigned long long)emb +
                              ((unsigned long long)patch0 * 1024u + (unsigned)kb * 32u) * 4u;
      v4u g0;
      g0[0] = 1u;                    // count = 1
      g0[1] = as_lds;                // LDS destination
      g0[2] = (unsigned int)ga;
      g0[3] = (unsigned int)((ga >> 32) & 0x1FFFFFFu) | (2u << 30); // type=2
      asm volatile("tensor_load_to_lds %0, %1" :: "s"(g0), "s"(g1) : "memory");
    }
    { // B chunk: contiguous 32KB of packed W1 tiles, async to LDS
      unsigned long long gbase = w1t_ga + (unsigned long long)kb * 32768u;
      #pragma unroll
      for (int i = 0; i < 8; ++i) {
        unsigned int off = (unsigned int)(tid + i * 256) * 16u;
        async_copy_b128(bs_lds + off, gbase + off);
      }
    }
    wait_asynccnt0();
    if (tid_u < 32) __builtin_amdgcn_s_wait_tensorcnt(0);
    __syncthreads();

    // A fragments: read f32 from padded LDS tile, convert to bf16
    v16bf a[2];
    #pragma unroll
    for (int m = 0; m < 2; ++m) {
      int row = wm * 32 + m * 16 + l16;
      const float* ap = Asf + row * 36 + half * 8;
      float4 x0 = ((const float4*)ap)[0];
      float4 x1 = ((const float4*)ap)[1];
      float4 x2 = ((const float4*)(ap + 16))[0];
      float4 x3 = ((const float4*)(ap + 16))[1];
      v16bf t;
      t[0]=(__bf16)x0.x; t[1]=(__bf16)x0.y; t[2]=(__bf16)x0.z; t[3]=(__bf16)x0.w;
      t[4]=(__bf16)x1.x; t[5]=(__bf16)x1.y; t[6]=(__bf16)x1.z; t[7]=(__bf16)x1.w;
      t[8]=(__bf16)x2.x; t[9]=(__bf16)x2.y; t[10]=(__bf16)x2.z; t[11]=(__bf16)x2.w;
      t[12]=(__bf16)x3.x; t[13]=(__bf16)x3.y; t[14]=(__bf16)x3.z; t[15]=(__bf16)x3.w;
      a[m] = t;
    }
    #pragma unroll
    for (int n = 0; n < 8; ++n) {
      int nb = wn * 8 + n;
      v16bf b = *(const v16bf*)(Bs + nb * 512 + lane * 16);
      #pragma unroll
      for (int m = 0; m < 2; ++m)
        acc[m][n] = __builtin_amdgcn_wmma_f32_16x16x32_bf16(
            false, a[m], false, b, (short)0, acc[m][n], false, false);
    }
  }

  __syncthreads(); // all A/B reads done before Hs (aliased) is written

  // epilogue 1: bias + exact GELU; spill f16 to HBM, bf16 tile to LDS
  #pragma unroll
  for (int m = 0; m < 2; ++m) {
    #pragma unroll
    for (int n = 0; n < 8; ++n) {
      int col = wn * 128 + n * 16 + l16;
      float bias = b1[col];
      #pragma unroll
      for (int p = 0; p < 8; ++p) {
        int row = wm * 32 + m * 16 + half * 8 + p;
        float v = gelu_exact(acc[m][n][p] + bias);
        Hs[row * 528 + col] = (__bf16)v;
        hout[(size_t)(patch0 + row) * 512u + col] = (_Float16)v;
      }
    }
  }
  if (tid < 64) s_score[tid] = 0.f;

  // ----------------- phase 2: h @ Wa1 -----------------
  v8f acc2[2][4];
  #pragma unroll
  for (int m = 0; m < 2; ++m)
    #pragma unroll
    for (int n = 0; n < 4; ++n) acc2[m][n] = vzero;

  for (int kb = 0; kb < 16; ++kb) {
    __syncthreads();
    { // Wa1 chunk: 16KB contiguous, async to LDS
      unsigned long long gbase = wa1t_ga + (unsigned long long)kb * 16384u;
      #pragma unroll
      for (int i = 0; i < 4; ++i) {
        unsigned int off = (unsigned int)(tid + i * 256) * 16u;
        async_copy_b128(w2s_lds + off, gbase + off);
      }
    }
    wait_asynccnt0();
    __syncthreads();

    v16bf a[2];
    #pragma unroll
    for (int m = 0; m < 2; ++m) {
      int row = wm * 32 + m * 16 + l16;
      typedef __attribute__((ext_vector_type(8))) __bf16 v8bf_t;
      v8bf_t lo = *(const v8bf_t*)(Hs + row * 528 + kb * 32 + half * 8);
      v8bf_t hi = *(const v8bf_t*)(Hs + row * 528 + kb * 32 + half * 8 + 16);
      a[m] = __builtin_shufflevector(lo, hi, 0,1,2,3,4,5,6,7,8,9,10,11,12,13,14,15);
    }
    #pragma unroll
    for (int n = 0; n < 4; ++n) {
      int nb = wn * 4 + n;
      v16bf b = *(const v16bf*)(W2s + nb * 512 + lane * 16);
      #pragma unroll
      for (int m = 0; m < 2; ++m)
        acc2[m][n] = __builtin_amdgcn_wmma_f32_16x16x32_bf16(
            false, a[m], false, b, (short)0, acc2[m][n], false, false);
    }
  }

  // epilogue 2: tanh(.)*Wa2, reduce rows across lanes + N-waves
  #pragma unroll
  for (int m = 0; m < 2; ++m) {
    float rs[8] = {0.f, 0.f, 0.f, 0.f, 0.f, 0.f, 0.f, 0.f};
    #pragma unroll
    for (int n = 0; n < 4; ++n) {
      int col = wn * 64 + n * 16 + l16;
      float bia = ba1[col];
      float w2  = Wa2[col];
      #pragma unroll
      for (int p = 0; p < 8; ++p)
        rs[p] += tanhf(acc2[m][n][p] + bia) * w2;
    }
    #pragma unroll
    for (int p = 0; p < 8; ++p) {
      float v = rs[p];
      v += __shfl_xor(v, 1, 32);
      v += __shfl_xor(v, 2, 32);
      v += __shfl_xor(v, 4, 32);
      v += __shfl_xor(v, 8, 32);
      if (l16 == 0) {
        int row = wm * 32 + m * 16 + half * 8 + p;
        atomicAdd(&s_score[row], v);
      }
    }
  }
  __syncthreads();
  if (tid < 64) scores[patch0 + tid] = s_score[tid] + ba2[0];
}

// ---------------------------------------------------------------------------
// Kernel 2: per-region softmax stats (max, sum exp) + zero region features
// region r owns patches {r + 16*m}, m = 0..4095
// ---------------------------------------------------------------------------
__global__ void region_stats(const float* __restrict__ scores,
                             float* __restrict__ stats, float* __restrict__ rf) {
  __shared__ float red[256];
  int r = blockIdx.x, t = threadIdx.x;
  float mx = -3.4e38f;
  for (int m = t; m < 4096; m += 256) mx = fmaxf(mx, scores[r + (m << 4)]);
  red[t] = mx; __syncthreads();
  for (int s = 128; s > 0; s >>= 1) { if (t < s) red[t] = fmaxf(red[t], red[t + s]); __syncthreads(); }
  mx = red[0]; __syncthreads();
  float sum = 0.f;
  for (int m = t; m < 4096; m += 256) sum += expf(scores[r + (m << 4)] - mx);
  red[t] = sum; __syncthreads();
  for (int s = 128; s > 0; s >>= 1) { if (t < s) red[t] += red[t + s]; __syncthreads(); }
  if (t == 0) { stats[2 * r] = mx; stats[2 * r + 1] = red[0]; }
  rf[r * 512 + t] = 0.f;
  rf[r * 512 + 256 + t] = 0.f;
}

// ---------------------------------------------------------------------------
// Kernel 3: region_features[r] = sum_m softmax(s)[r,m] * h[r+16m]
// ---------------------------------------------------------------------------
__global__ void region_pool(const _Float16* __restrict__ h,
                            const float* __restrict__ scores,
                            const float* __restrict__ stats,
                            float* __restrict__ rf) {
  int b  = blockIdx.x;
  int r  = b >> 5;
  int cc = (b >> 3) & 3;
  int mc = b & 7;
  int t  = threadIdx.x; // 128 threads
  int col = cc * 128 + t;
  float mx = stats[2 * r], invZ = 1.f / stats[2 * r + 1];
  float acc = 0.f;
  int m0 = mc * 512;
  #pragma unroll 4
  for (int m = m0; m < m0 + 512; ++m) {
    int patch = r + (m << 4);
    float w = expf(scores[patch] - mx);
    acc += w * (float)h[(size_t)patch * 512u + col];
  }
  atomicAdd(&rf[r * 512 + col], acc * invZ);
}

// ---------------------------------------------------------------------------
// Kernel 4a: slide_emb = gelu(rf @ Ws + bs)
// ---------------------------------------------------------------------------
__global__ void slide_enc(const float* __restrict__ rf, const float* __restrict__ Ws,
                          const float* __restrict__ bs, float* __restrict__ semb) {
  __shared__ float row[512];
  int r  = blockIdx.x >> 1;
  int gh = (blockIdx.x & 1) * 256 + threadIdx.x;
  row[threadIdx.x]       = rf[r * 512 + threadIdx.x];
  row[256 + threadIdx.x] = rf[r * 512 + 256 + threadIdx.x];
  __syncthreads();
  float acc = bs[gh];
  for (int k = 0; k < 512; ++k) acc += row[k] * Ws[k * 512 + gh];
  semb[r * 512 + gh] = gelu_exact(acc);
}

// ---------------------------------------------------------------------------
// Kernel 4b: sattn[r] = Wsa2 . tanh(slide_emb[r] @ Wsa1 + bsa1) + bsa2
// ---------------------------------------------------------------------------
__global__ void slide_attn(const float* __restrict__ semb, const float* __restrict__ Wsa1,
                           const float* __restrict__ bsa1, const float* __restrict__ Wsa2,
                           const float* __restrict__ bsa2, float* __restrict__ sattn) {
  __shared__ float row[512];
  __shared__ float red[256];
  int r = blockIdx.x, t = threadIdx.x;
  row[t]       = semb[r * 512 + t];
  row[256 + t] = semb[r * 512 + 256 + t];
  __syncthreads();
  float acc = bsa1[t];
  for (int k = 0; k < 512; ++k) acc += row[k] * Wsa1[k * 256 + t];
  red[t] = tanhf(acc) * Wsa2[t];
  __syncthreads();
  for (int s = 128; s > 0; s >>= 1) { if (t < s) red[t] += red[t + s]; __syncthreads(); }
  if (t == 0) sattn[r] = red[0] + bsa2[0];
}

// ---------------------------------------------------------------------------
// Kernel 4c: softmax over 16 regions, slide_rep, classifier -> logits (2)
// ---------------------------------------------------------------------------
__global__ void head_kernel(const float* __restrict__ semb, const float* __restrict__ sattn,
                            const float* __restrict__ Wc1, const float* __restrict__ bc1,
                            const float* __restrict__ Wc2, const float* __restrict__ bc2,
                            float* __restrict__ out) {
  __shared__ float w[16];
  __shared__ float rep[512];
  __shared__ float hid[256];
  int t = threadIdx.x;
  if (t == 0) {
    float mx = -3.4e38f;
    for (int r = 0; r < 16; ++r) mx = fmaxf(mx, sattn[r]);
    float e[16], Z = 0.f;
    for (int r = 0; r < 16; ++r) { e[r] = expf(sattn[r] - mx); Z += e[r]; }
    for (int r = 0; r < 16; ++r) w[r] = e[r] / Z;
  }
  __syncthreads();
  for (int h = t; h < 512; h += 256) {
    float a = 0.f;
    for (int r = 0; r < 16; ++r) a += w[r] * semb[r * 512 + h];
    rep[h] = a;
  }
  __syncthreads();
  {
    float a = bc1[t];
    for (int k = 0; k < 512; ++k) a += rep[k] * Wc1[k * 256 + t];
    hid[t] = gelu_exact(a);
  }
  __syncthreads();
  if (t < 2) {
    float a = bc2[t];
    for (int j = 0; j < 256; ++j) a += hid[j] * Wc2[j * 2 + t];
    out[t] = a;
  }
}

// ---------------------------------------------------------------------------
extern "C" void kernel_launch(void* const* d_in, const int* in_sizes, int n_in,
                              void* d_out, int out_size, void* d_ws, size_t ws_size,
                              hipStream_t stream) {
  const float* emb  = (const float*)d_in[0];
  const float* W1   = (const float*)d_in[1];
  const float* b1   = (const float*)d_in[2];
  const float* Wa1  = (const float*)d_in[3];
  const float* ba1  = (const float*)d_in[4];
  const float* Wa2  = (const float*)d_in[5];
  const float* ba2  = (const float*)d_in[6];
  const float* Ws   = (const float*)d_in[7];
  const float* bs   = (const float*)d_in[8];
  const float* Wsa1 = (const float*)d_in[9];
  const float* bsa1 = (const float*)d_in[10];
  const float* Wsa2 = (const float*)d_in[11];
  const float* bsa2 = (const float*)d_in[12];
  const float* Wc1  = (const float*)d_in[13];
  const float* bc1  = (const float*)d_in[14];
  const float* Wc2  = (const float*)d_in[15];
  const float* bc2  = (const float*)d_in[16];

  char* ws = (char*)d_ws;
  size_t off = 0;
  auto carve = [&](size_t bytes) -> void* {
    void* p = ws + off;
    off = (off + bytes + 255) & ~(size_t)255;
    return p;
  };
  __bf16*    W1t    = (__bf16*)   carve((size_t)1024 * 512 * 2); // 1 MB
  __bf16*    Wa1t   = (__bf16*)   carve((size_t)512 * 256 * 2);  // 256 KB
  _Float16*  hbuf   = (_Float16*) carve((size_t)65536 * 512 * 2);// 64 MB
  float*     scores = (float*)    carve((size_t)65536 * 4);      // 256 KB
  float*     stats  = (float*)    carve((size_t)32 * 4);
  float*     rf     = (float*)    carve((size_t)16 * 512 * 4);
  float*     semb   = (float*)    carve((size_t)16 * 512 * 4);
  float*     sattn  = (float*)    carve((size_t)16 * 4);

  pack_b_bf16<<<2048, 256, 0, stream>>>(W1, W1t, 1024, 512);
  pack_b_bf16<<<512, 256, 0, stream>>>(Wa1, Wa1t, 512, 256);

  mil_stage1<<<1024, 256, 0, stream>>>(emb, W1t, b1, Wa1t, ba1, Wa2, ba2, hbuf, scores);

  region_stats<<<16, 256, 0, stream>>>(scores, stats, rf);
  region_pool<<<512, 128, 0, stream>>>(hbuf, scores, stats, rf);

  slide_enc<<<32, 256, 0, stream>>>(rf, Ws, bs, semb);
  slide_attn<<<16, 256, 0, stream>>>(semb, Wsa1, bsa1, Wsa2, bsa2, sattn);
  head_kernel<<<1, 256, 0, stream>>>(semb, sattn, Wc1, bc1, Wc2, bc2, (float*)d_out);
}
